// NERNetCombined2_59038620451314
// MI455X (gfx1250) — matrix operand order repaired and checked by reading, not compiled, not run on hardware
//
#include <hip/hip_runtime.h>

// ---------------------------------------------------------------------------
// Types for CDNA5 WMMA (wave32, 16x16x32 bf16 -> f32)
// ---------------------------------------------------------------------------
typedef __attribute__((ext_vector_type(16))) __bf16 bf16x16;
typedef __attribute__((ext_vector_type(8)))  float  floatx8;
typedef __attribute__((ext_vector_type(4)))  unsigned int u32x4;

union TileU { bf16x16 v; u32x4 q[2]; };

__device__ __forceinline__ floatx8 zero8() {
  floatx8 v;
  for (int i = 0; i < 8; ++i) v[i] = 0.f;
  return v;
}

__device__ __forceinline__ unsigned short f2bf(float f) {
  union { float f; unsigned u; } x; x.f = f;
  unsigned u = x.u;
  u += 0x7fffu + ((u >> 16) & 1u);   // round-to-nearest-even
  return (unsigned short)(u >> 16);
}

// A-matrix 16x32 bf16 tile (M x K), row-major source, row stride ld (elems).
__device__ __forceinline__ bf16x16 load_tile_a(const unsigned short* base, int ld) {
  int lane = threadIdx.x & 31;
  int l = lane & 15, hi = lane >> 4;
  const unsigned short* p = base + l * ld + hi * 8;
  TileU u;
  u.q[0] = *(const u32x4*)(p);
  u.q[1] = *(const u32x4*)(p + 16);
  return u.v;
}

// B-matrix 32x16 bf16 tile (K x N), row-major source, row stride ld (elems).
__device__ __forceinline__ bf16x16 load_tile_b(const unsigned short* base, int ld) {
  int lane = threadIdx.x & 31;
  int l = lane & 15, hi = lane >> 4;
  const unsigned short* p = base + l;
  unsigned short tmp[16];
  for (int s = 0; s < 8; ++s) {
    tmp[s] = p[(size_t)(hi * 8 + s) * ld];
    tmp[8 + s] = p[(size_t)(16 + hi * 8 + s) * ld];
  }
  TileU u;
  __builtin_memcpy(&u, tmp, sizeof(tmp));
  return u.v;
}

__device__ __forceinline__ floatx8 wmma_bf16(bf16x16 a, bf16x16 b, floatx8 c) {
  return __builtin_amdgcn_wmma_f32_16x16x32_bf16(false, a, false, b, (short)0, c,
                                                 false, false);
}

__device__ __forceinline__ float sigm(float x) { return 1.f / (1.f + __expf(-x)); }

// Problem constants
#define BB 32
#define SS 256
#define CC 16
#define EPAD 480     // E=460 padded to 480 -> 15 k-steps of 32
#define HH 512
#define G4H 2048
#define D2H 1024
#define TT 17

// ---------------------------------------------------------------------------
// 1) Embedding gather -> feats bf16 [B*S][480] (zero padded 460..479)
// ---------------------------------------------------------------------------
__global__ void embed_kernel(const int* __restrict__ xt, const int* __restrict__ xp,
                             const int* __restrict__ xc, const int* __restrict__ xC,
                             const int* __restrict__ xn,
                             const float* __restrict__ tokens, const float* __restrict__ tokenp,
                             const float* __restrict__ pos, const float* __restrict__ chars,
                             const float* __restrict__ cap, const float* __restrict__ num,
                             unsigned short* __restrict__ feats) {
  int bs = blockIdx.x;
  int ti = xt[bs], pi = xp[bs], ci = xC[bs], ni = xn[bs];
  for (int e = threadIdx.x; e < EPAD; e += blockDim.x) {
    float v = 0.f;
    if (e < 300) {
      if (ti >= 10000) v = tokenp[(size_t)(ti - 10000) * 300 + e];
      else             v = tokens[(size_t)ti * 300 + e];
    } else if (e < 364) {
      v = pos[(size_t)pi * 64 + (e - 300)];
    } else if (e < 428) {
      int d = e - 364;
      float s = 0.f;
      for (int c = 0; c < CC; ++c) s += chars[(size_t)xc[bs * CC + c] * 64 + d];
      v = s * (1.f / 16.f);
    } else if (e < 444) {
      v = cap[ci * 16 + (e - 428)];
    } else if (e < 460) {
      v = num[ni * 16 + (e - 444)];
    }
    feats[(size_t)bs * EPAD + e] = f2bf(v);
  }
}

// ---------------------------------------------------------------------------
// 2) Weight transpose + bf16 conversion kernels
// ---------------------------------------------------------------------------
__global__ void conv_wih_kernel(const float* __restrict__ wf, const float* __restrict__ wb,
                                unsigned short* __restrict__ out) {
  int i = blockIdx.x * blockDim.x + threadIdx.x;
  if (i >= 2 * EPAD * G4H) return;
  int dir = i / (EPAD * G4H);
  int rem = i % (EPAD * G4H);
  int k = rem / G4H, n = rem % G4H;
  const float* w = dir ? wb : wf;
  out[i] = (k < 460) ? f2bf(w[(size_t)n * 460 + k]) : (unsigned short)0;
}

__global__ void conv_whh_kernel(const float* __restrict__ wf, const float* __restrict__ wb,
                                unsigned short* __restrict__ out) {
  int i = blockIdx.x * blockDim.x + threadIdx.x;
  if (i >= 2 * HH * G4H) return;
  int dir = i / (HH * G4H);
  int rem = i % (HH * G4H);
  int k = rem / G4H, n = rem % G4H;
  const float* w = dir ? wb : wf;
  out[i] = f2bf(w[(size_t)n * HH + k]);
}

// ---------------------------------------------------------------------------
// 3) Input projection GEMM: xg[dir][B*S][2048] = feats @ wihT[dir] + b_ih + b_hh
//    2x2 register blocking: each wave computes 32x32 (4 WMMA tiles), sharing
//    A across n-tiles and B across m-tiles -> 1 b128 A load per WMMA; B via 16-bit loads.
//    block = 128 threads (4 waves); grid (256, 16, 2)
// ---------------------------------------------------------------------------
__global__ __launch_bounds__(128) void xg_gemm_kernel(
    const unsigned short* __restrict__ feats, const unsigned short* __restrict__ wihT,
    const float* __restrict__ bihf, const float* __restrict__ bhhf,
    const float* __restrict__ bihb, const float* __restrict__ bhhb,
    float* __restrict__ xg) {
  int dir = blockIdx.z;
  int m0 = blockIdx.x * 32;
  int n0 = (blockIdx.y * 4 + (threadIdx.x >> 5)) * 32;
  const unsigned short* wd = wihT + (size_t)dir * EPAD * G4H;
  const float* bi = dir ? bihb : bihf;
  const float* bh = dir ? bhhb : bhhf;
  floatx8 a00 = zero8(), a01 = zero8(), a10 = zero8(), a11 = zero8();
  for (int ks = 0; ks < EPAD / 32; ++ks) {
    bf16x16 A0 = load_tile_a(feats + (size_t)m0 * EPAD + ks * 32, EPAD);
    bf16x16 A1 = load_tile_a(feats + (size_t)(m0 + 16) * EPAD + ks * 32, EPAD);
    bf16x16 B0 = load_tile_b(wd + (size_t)(ks * 32) * G4H + n0, G4H);
    bf16x16 B1 = load_tile_b(wd + (size_t)(ks * 32) * G4H + n0 + 16, G4H);
    a00 = wmma_bf16(A0, B0, a00);
    a01 = wmma_bf16(A0, B1, a01);
    a10 = wmma_bf16(A1, B0, a10);
    a11 = wmma_bf16(A1, B1, a11);
  }
  int lane = threadIdx.x & 31, l = lane & 15, hi = lane >> 4;
  float* xd = xg + (size_t)dir * (BB * SS) * G4H;
  int c0 = n0 + l, c1 = n0 + 16 + l;
  float bias0 = bi[c0] + bh[c0];
  float bias1 = bi[c1] + bh[c1];
  for (int r = 0; r < 8; ++r) {
    int r0 = m0 + hi * 8 + r;
    int r1 = m0 + 16 + hi * 8 + r;
    xd[(size_t)r0 * G4H + c0] = a00[r] + bias0;
    xd[(size_t)r0 * G4H + c1] = a01[r] + bias1;
    xd[(size_t)r1 * G4H + c0] = a10[r] + bias0;
    xd[(size_t)r1 * G4H + c1] = a11[r] + bias1;
  }
}

// ---------------------------------------------------------------------------
// 4) BiLSTM recurrence. One persistent block per direction (512 thr = 16 waves).
//    h in LDS bf16 (ping-pong), c in VGPRs, w_hhT streamed from L2.
//    A tile shared across the 4 gate WMMAs.
// ---------------------------------------------------------------------------
__global__ __launch_bounds__(512) void lstm_kernel(
    const float* __restrict__ xg, const unsigned short* __restrict__ whhT,
    const int* __restrict__ lens, unsigned short* __restrict__ hcomb) {
  int dir = blockIdx.x;
  const float* xgd = xg + (size_t)dir * (BB * SS) * G4H;
  const unsigned short* wt = whhT + (size_t)dir * HH * G4H;

  __shared__ alignas(16) unsigned short hbuf[2][BB * HH];  // 64 KB ping-pong
  int tid = threadIdx.x;
  {
    unsigned short* hz = &hbuf[0][0];
    for (int i = tid; i < 2 * BB * HH; i += 512) hz[i] = 0;
  }
  __syncthreads();

  int wave = tid >> 5;
  int lane = tid & 31, l = lane & 15, hi = lane >> 4;

  int lenc[2][8];
  for (int mt = 0; mt < 2; ++mt)
    for (int r = 0; r < 8; ++r) lenc[mt][r] = lens[mt * 16 + hi * 8 + r];

  floatx8 cst[4];
  for (int p = 0; p < 4; ++p) cst[p] = zero8();

  int src = 0;
  for (int t = 0; t < SS; ++t) {
    unsigned short* hs = hbuf[src];
    unsigned short* hd = hbuf[src ^ 1];
    for (int p = 0; p < 4; ++p) {
      int mt = p >> 1;
      int ut = wave + (p & 1) * 16;    // unit tile 0..31
      int n0 = ut * 16;
      floatx8 g0 = zero8(), g1 = zero8(), g2 = zero8(), g3 = zero8();
      for (int ks = 0; ks < HH / 32; ++ks) {
        bf16x16 a = load_tile_a(hs + (mt * 16) * HH + ks * 32, HH);
        const unsigned short* bbp = wt + (size_t)(ks * 32) * G4H + n0;
        g0 = wmma_bf16(a, load_tile_b(bbp + 0 * HH, G4H), g0);
        g1 = wmma_bf16(a, load_tile_b(bbp + 1 * HH, G4H), g1);
        g2 = wmma_bf16(a, load_tile_b(bbp + 2 * HH, G4H), g2);
        g3 = wmma_bf16(a, load_tile_b(bbp + 3 * HH, G4H), g3);
      }
      for (int r = 0; r < 8; ++r) {
        int bI = mt * 16 + hi * 8 + r;
        int len = lenc[mt][r];
        int pos = (dir == 0) ? t : ((t < len) ? (len - 1 - t) : t);
        size_t xrow = (((size_t)bI) * SS + pos) * G4H;
        int col = n0 + l;
        float gi = g0[r] + xgd[xrow + 0 * HH + col];
        float gf = g1[r] + xgd[xrow + 1 * HH + col];
        float gg = g2[r] + xgd[xrow + 2 * HH + col];
        float go = g3[r] + xgd[xrow + 3 * HH + col];
        float c = sigm(gf) * cst[p][r] + sigm(gi) * tanhf(gg);
        cst[p][r] = c;
        float h = sigm(go) * tanhf(c);
        hd[bI * HH + col] = f2bf(h);                       // state for next step
        float hm = (t < len) ? h : 0.f;                    // output mask
        hcomb[(((size_t)bI) * SS + pos) * D2H + dir * HH + col] = f2bf(hm);
      }
    }
    __syncthreads();
    src ^= 1;
  }
}

// ---------------------------------------------------------------------------
// 5) Transpose h -> hT [B][1024][256] bf16 (contiguous B-tiles for scores)
// ---------------------------------------------------------------------------
__global__ void transpose_h_kernel(const unsigned short* __restrict__ hcomb,
                                   unsigned short* __restrict__ hT) {
  int i = blockIdx.x * blockDim.x + threadIdx.x;
  if (i >= BB * D2H * SS) return;
  int b = i / (D2H * SS);
  int rem = i % (D2H * SS);
  int d = rem / SS, s = rem % SS;
  hT[i] = hcomb[((size_t)b * SS + s) * D2H + d];
}

// ---------------------------------------------------------------------------
// 6) Attention scores: S[b] = h[b] @ h[b]^T (256x1024 x 1024x256), 2x2 blocked
//    grid (8, 2, 32), block 128
// ---------------------------------------------------------------------------
__global__ __launch_bounds__(128) void scores_kernel(
    const unsigned short* __restrict__ hcomb, const unsigned short* __restrict__ hT,
    float* __restrict__ scores) {
  int b = blockIdx.z;
  int m0 = blockIdx.x * 32;
  int n0 = (blockIdx.y * 4 + (threadIdx.x >> 5)) * 32;
  const unsigned short* A = hcomb + (size_t)b * SS * D2H;
  const unsigned short* Bm = hT + (size_t)b * D2H * SS;
  floatx8 a00 = zero8(), a01 = zero8(), a10 = zero8(), a11 = zero8();
  for (int ks = 0; ks < D2H / 32; ++ks) {
    bf16x16 A0 = load_tile_a(A + (size_t)m0 * D2H + ks * 32, D2H);
    bf16x16 A1 = load_tile_a(A + (size_t)(m0 + 16) * D2H + ks * 32, D2H);
    bf16x16 B0 = load_tile_b(Bm + (size_t)(ks * 32) * SS + n0, SS);
    bf16x16 B1 = load_tile_b(Bm + (size_t)(ks * 32) * SS + n0 + 16, SS);
    a00 = wmma_bf16(A0, B0, a00);
    a01 = wmma_bf16(A0, B1, a01);
    a10 = wmma_bf16(A1, B0, a10);
    a11 = wmma_bf16(A1, B1, a11);
  }
  int lane = threadIdx.x & 31, l = lane & 15, hi = lane >> 4;
  float* Sb = scores + (size_t)b * SS * SS;
  for (int r = 0; r < 8; ++r) {
    int r0 = m0 + hi * 8 + r;
    int r1 = m0 + 16 + hi * 8 + r;
    Sb[(size_t)r0 * SS + n0 + l]      = a00[r];
    Sb[(size_t)r0 * SS + n0 + 16 + l] = a01[r];
    Sb[(size_t)r1 * SS + n0 + l]      = a10[r];
    Sb[(size_t)r1 * SS + n0 + 16 + l] = a11[r];
  }
}

// ---------------------------------------------------------------------------
// 7) Row softmax (over all 256 positions; matches reference, no -inf mask)
// ---------------------------------------------------------------------------
__global__ void softmax_kernel(const float* __restrict__ scores,
                               unsigned short* __restrict__ abf) {
  int row = blockIdx.x;           // b*256 + s
  int tid = threadIdx.x;
  __shared__ float buf[256];
  float x = scores[(size_t)row * SS + tid];
  buf[tid] = x; __syncthreads();
  for (int s = 128; s > 0; s >>= 1) {
    if (tid < s) buf[tid] = fmaxf(buf[tid], buf[tid + s]);
    __syncthreads();
  }
  float mx = buf[0]; __syncthreads();
  float e = __expf(x - mx);
  buf[tid] = e; __syncthreads();
  for (int s = 128; s > 0; s >>= 1) {
    if (tid < s) buf[tid] += buf[tid + s];
    __syncthreads();
  }
  float sm = buf[0];
  abf[(size_t)row * SS + tid] = f2bf(e / sm);
}

// ---------------------------------------------------------------------------
// 8) h2[b] = a[b] @ h[b] (256x256 x 256x1024) -> f32, 2x2 blocked
//    grid (8, 8, 32), block 128
// ---------------------------------------------------------------------------
__global__ __launch_bounds__(128) void av_kernel(
    const unsigned short* __restrict__ abf, const unsigned short* __restrict__ hcomb,
    float* __restrict__ h2) {
  int b = blockIdx.z;
  int m0 = blockIdx.x * 32;
  int n0 = (blockIdx.y * 4 + (threadIdx.x >> 5)) * 32;
  const unsigned short* A = abf + (size_t)b * SS * SS;
  const unsigned short* Bm = hcomb + (size_t)b * SS * D2H;
  floatx8 a00 = zero8(), a01 = zero8(), a10 = zero8(), a11 = zero8();
  for (int ks = 0; ks < SS / 32; ++ks) {
    bf16x16 A0 = load_tile_a(A + (size_t)m0 * SS + ks * 32, SS);
    bf16x16 A1 = load_tile_a(A + (size_t)(m0 + 16) * SS + ks * 32, SS);
    bf16x16 B0 = load_tile_b(Bm + (size_t)(ks * 32) * D2H + n0, D2H);
    bf16x16 B1 = load_tile_b(Bm + (size_t)(ks * 32) * D2H + n0 + 16, D2H);
    a00 = wmma_bf16(A0, B0, a00);
    a01 = wmma_bf16(A0, B1, a01);
    a10 = wmma_bf16(A1, B0, a10);
    a11 = wmma_bf16(A1, B1, a11);
  }
  int lane = threadIdx.x & 31, l = lane & 15, hi = lane >> 4;
  float* Ob = h2 + (size_t)b * SS * D2H;
  for (int r = 0; r < 8; ++r) {
    int r0 = m0 + hi * 8 + r;
    int r1 = m0 + 16 + hi * 8 + r;
    Ob[(size_t)r0 * D2H + n0 + l]      = a00[r];
    Ob[(size_t)r0 * D2H + n0 + 16 + l] = a01[r];
    Ob[(size_t)r1 * D2H + n0 + l]      = a10[r];
    Ob[(size_t)r1 * D2H + n0 + 16 + l] = a11[r];
  }
}

// ---------------------------------------------------------------------------
// 9) Emissions: em[bs][t] = h2[bs] . fc_w[t] + fc_b[t]   (T=17, VALU dot)
// ---------------------------------------------------------------------------
__global__ void em_kernel(const float* __restrict__ h2, const float* __restrict__ fcw,
                          const float* __restrict__ fcb, float* __restrict__ em) {
  int idx = blockIdx.x * blockDim.x + threadIdx.x;
  if (idx >= BB * SS * TT) return;
  int t = idx % TT;
  int bs = idx / TT;
  const float* hrow = h2 + (size_t)bs * D2H;
  const float* wrow = fcw + (size_t)t * D2H;
  float s = 0.f;
#pragma unroll 4
  for (int k = 0; k < D2H; ++k) s += hrow[k] * wrow[k];
  em[idx] = s + fcb[t];
}

// ---------------------------------------------------------------------------
// 10) loss init, CRF forward (NLL) and Viterbi. One wave per batch element.
// ---------------------------------------------------------------------------
__global__ void init_loss_kernel(float* p) { *p = 0.f; }

__global__ void crf_nll_kernel(const float* __restrict__ em, const int* __restrict__ y,
                               const int* __restrict__ lens, const float* __restrict__ start,
                               const float* __restrict__ end_, const float* __restrict__ trans,
                               float* __restrict__ loss_out) {
  int b = blockIdx.x;
  int j = threadIdx.x;
  __shared__ float alpha[32];
  __shared__ float tr[TT * TT];
  int len = lens[b];
  for (int i = j; i < TT * TT; i += 32) tr[i] = trans[i];
  if (j < TT) alpha[j] = start[j] + em[((size_t)b * SS) * TT + j];
  __syncthreads();
  for (int t = 1; t < SS; ++t) {
    float nv = 0.f;
    if (j < TT) {
      float mx = -1e30f;
      for (int i = 0; i < TT; ++i) mx = fmaxf(mx, alpha[i] + tr[i * TT + j]);
      float s = 0.f;
      for (int i = 0; i < TT; ++i) s += __expf(alpha[i] + tr[i * TT + j] - mx);
      nv = mx + __logf(s) + em[((size_t)b * SS + t) * TT + j];
      if (t >= len) nv = alpha[j];
    }
    __syncthreads();
    if (j < TT) alpha[j] = nv;
    __syncthreads();
  }
  if (j == 0) {
    float mx = -1e30f;
    for (int i = 0; i < TT; ++i) mx = fmaxf(mx, alpha[i] + end_[i]);
    float s = 0.f;
    for (int i = 0; i < TT; ++i) s += __expf(alpha[i] + end_[i] - mx);
    float logZ = mx + __logf(s);
    const int* yb = y + b * SS;
    float gold = start[yb[0]] + em[((size_t)b * SS) * TT + yb[0]];
    for (int t = 1; t < SS; ++t)
      if (t < len) gold += tr[yb[t - 1] * TT + yb[t]] + em[((size_t)b * SS + t) * TT + yb[t]];
    gold += end_[yb[len - 1]];
    atomicAdd(loss_out, logZ - gold);   // loss = -sum(gold - logZ)
  }
}

__global__ void viterbi_kernel(const float* __restrict__ em, const int* __restrict__ lens,
                               const float* __restrict__ start, const float* __restrict__ end_,
                               const float* __restrict__ trans,
                               unsigned char* __restrict__ hist, float* __restrict__ out_tags) {
  int b = blockIdx.x;
  int j = threadIdx.x;
  __shared__ float sc[32];
  __shared__ float tr[TT * TT];
  int len = lens[b];
  for (int i = j; i < TT * TT; i += 32) tr[i] = trans[i];
  if (j < TT) sc[j] = start[j] + em[((size_t)b * SS) * TT + j];
  __syncthreads();
  unsigned char* hb = hist + (size_t)b * (SS - 1) * TT;
  for (int t = 1; t < SS; ++t) {
    float nv = 0.f;
    if (j < TT) {
      float best = -1e30f; int arg = 0;
      for (int i = 0; i < TT; ++i) {
        float v = sc[i] + tr[i * TT + j];
        if (v > best) { best = v; arg = i; }
      }
      hb[(t - 1) * TT + j] = (unsigned char)arg;
      nv = best + em[((size_t)b * SS + t) * TT + j];
      if (t >= len) nv = sc[j];
    }
    __syncthreads();
    if (j < TT) sc[j] = nv;
    __syncthreads();
  }
  if (j == 0) {
    float best = -1e30f; int cur = 0;
    for (int i = 0; i < TT; ++i) {
      float v = sc[i] + end_[i];
      if (v > best) { best = v; cur = i; }
    }
    float* ob = out_tags + b * SS;
    ob[SS - 1] = ((SS - 1) < len) ? (float)cur : 0.f;
    for (int t = SS - 1; t >= 1; --t) {
      if (t < len) cur = hb[(t - 1) * TT + cur];
      ob[t - 1] = ((t - 1) < len) ? (float)cur : 0.f;
    }
  }
}

// ---------------------------------------------------------------------------
// Launch
// ---------------------------------------------------------------------------
extern "C" void kernel_launch(void* const* d_in, const int* in_sizes, int n_in,
                              void* d_out, int out_size, void* d_ws, size_t ws_size,
                              hipStream_t stream) {
  (void)in_sizes; (void)n_in; (void)out_size; (void)ws_size;
  const int*   xt     = (const int*)d_in[0];
  const int*   xp     = (const int*)d_in[1];
  const int*   xc     = (const int*)d_in[2];
  const int*   xCc    = (const int*)d_in[3];
  const int*   xn     = (const int*)d_in[4];
  const int*   lens   = (const int*)d_in[5];
  const int*   y      = (const int*)d_in[6];
  const float* tokens = (const float*)d_in[7];
  const float* tokenp = (const float*)d_in[8];
  const float* pos    = (const float*)d_in[9];
  const float* chars  = (const float*)d_in[10];
  const float* cap    = (const float*)d_in[11];
  const float* num    = (const float*)d_in[12];
  const float* wihf   = (const float*)d_in[13];
  const float* whhf   = (const float*)d_in[14];
  const float* bihf   = (const float*)d_in[15];
  const float* bhhf   = (const float*)d_in[16];
  const float* wihb   = (const float*)d_in[17];
  const float* whhb   = (const float*)d_in[18];
  const float* bihb   = (const float*)d_in[19];
  const float* bhhb   = (const float*)d_in[20];
  const float* fcw    = (const float*)d_in[21];
  const float* fcb    = (const float*)d_in[22];
  const float* crfs   = (const float*)d_in[23];
  const float* crfe   = (const float*)d_in[24];
  const float* crft   = (const float*)d_in[25];

  float* out = (float*)d_out;          // [0..8191] tags (as float), [8192] loss
  char*  ws  = (char*)d_ws;
  auto al = [](size_t x) { return (x + 255) & ~(size_t)255; };
  size_t o = 0;
  unsigned short* feats = (unsigned short*)(ws + o); o += al((size_t)BB * SS * EPAD * 2);
  unsigned short* wihT  = (unsigned short*)(ws + o); o += al((size_t)2 * EPAD * G4H * 2);
  unsigned short* whhT  = (unsigned short*)(ws + o); o += al((size_t)2 * HH * G4H * 2);
  float*          xg    = (float*)(ws + o);          o += al((size_t)2 * BB * SS * G4H * 4);
  unsigned short* hcomb = (unsigned short*)(ws + o); o += al((size_t)BB * SS * D2H * 2);
  unsigned short* hT    = (unsigned short*)(ws + o); o += al((size_t)BB * D2H * SS * 2);
  float*          scor  = (float*)(ws + o);          o += al((size_t)BB * SS * SS * 4);
  unsigned short* abf   = (unsigned short*)(ws + o); o += al((size_t)BB * SS * SS * 2);
  float*          h2    = (float*)(ws + o);          o += al((size_t)BB * SS * D2H * 4);
  float*          em    = (float*)(ws + o);          o += al((size_t)BB * SS * TT * 4);
  unsigned char*  hist  = (unsigned char*)(ws + o);  o += al((size_t)BB * (SS - 1) * TT);

  embed_kernel<<<BB * SS, 128, 0, stream>>>(xt, xp, xc, xCc, xn, tokens, tokenp, pos,
                                            chars, cap, num, feats);
  conv_wih_kernel<<<(2 * EPAD * G4H + 255) / 256, 256, 0, stream>>>(wihf, wihb, wihT);
  conv_whh_kernel<<<(2 * HH * G4H + 255) / 256, 256, 0, stream>>>(whhf, whhb, whhT);
  xg_gemm_kernel<<<dim3(BB * SS / 32, G4H / 128, 2), 128, 0, stream>>>(
      feats, wihT, bihf, bhhf, bihb, bhhb, xg);
  lstm_kernel<<<2, 512, 0, stream>>>(xg, whhT, lens, hcomb);
  transpose_h_kernel<<<(BB * D2H * SS + 255) / 256, 256, 0, stream>>>(hcomb, hT);
  scores_kernel<<<dim3(SS / 32, SS / 128, BB), 128, 0, stream>>>(hcomb, hT, scor);
  softmax_kernel<<<BB * SS, 256, 0, stream>>>(scor, abf);
  av_kernel<<<dim3(SS / 32, D2H / 128, BB), 128, 0, stream>>>(abf, hcomb, h2);
  em_kernel<<<(BB * SS * TT + 127) / 128, 128, 0, stream>>>(h2, fcw, fcb, em);
  init_loss_kernel<<<1, 1, 0, stream>>>(out + BB * SS);
  crf_nll_kernel<<<BB, 32, 0, stream>>>(em, y, lens, crfs, crfe, crft, out + BB * SS);
  viterbi_kernel<<<BB, 32, 0, stream>>>(em, lens, crfs, crfe, crft, hist, out);
}
